// SCT_GAT_ogbarxiv_89292370083910
// MI455X (gfx1250) — matrix operand order repaired and checked by reading, not compiled, not run on hardware
//
#include <hip/hip_runtime.h>
#include <stdint.h>

#define NN      6144
#define NF      128
#define HIDN    128
#define NHEADS  8
#define NCLS    40
#define NCP     48            // classes padded to 3x16 WMMA tiles
#define XC      (HIDN*NHEADS) // 1024
#define ALPHA   0.2f
#define EPSB    1e-5f
#define SMOO    0.5f

#define LDS_STRIDE 132        // 128 + 4 pad: lanes16-31 offset by 4*132 -> +16 banks (conflict-free)

#if defined(__has_builtin)
#  if __has_builtin(__builtin_amdgcn_global_load_async_to_lds_b128) && \
      __has_builtin(__builtin_amdgcn_s_wait_asynccnt)
#    define USE_ASYNC 1
#  endif
#endif
#ifndef USE_ASYNC
#  define USE_ASYNC 0
#endif

typedef __bf16 bf16x16 __attribute__((ext_vector_type(16)));
typedef float  f32x8   __attribute__((ext_vector_type(8)));

// exact parameter types of the async-to-LDS builtins (from hipcc diagnostic):
// (v4i __device__*, v4i __shared__*, Imm offset, Imm cpol)
typedef int i32x4v __attribute__((vector_size(4 * sizeof(int))));
typedef __attribute__((address_space(1))) i32x4v* gvec4p;
typedef __attribute__((address_space(3))) i32x4v* lvec4p;

union ABu { bf16x16 v; uint32_t u[8]; };

__device__ __forceinline__ unsigned short f2bf(float f) {
  uint32_t u = __float_as_uint(f);
  return (unsigned short)((u + 0x7FFFu + ((u >> 16) & 1u)) >> 16); // RNE truncate
}
__device__ __forceinline__ uint32_t pack2(float a, float b) {
  return (uint32_t)f2bf(a) | ((uint32_t)f2bf(b) << 16);
}

// A-matrix 16x32 bf16 tile from row-major bf16 storage.
// lane<16: rows 0..15, K = k0+{0..7, 16..23}; lane>=16: same rows, K = k0+{8..15, 24..31}
__device__ __forceinline__ ABu loadA16(const unsigned short* base, int row, int ld,
                                       int k0, int lane) {
  ABu a;
  int koff = (lane >= 16) ? 8 : 0;
  const uint4* p0 = (const uint4*)(base + (size_t)row * ld + k0 + koff);
  const uint4* p1 = (const uint4*)(base + (size_t)row * ld + k0 + koff + 16);
  uint4 x0 = *p0, x1 = *p1;
  a.u[0] = x0.x; a.u[1] = x0.y; a.u[2] = x0.z; a.u[3] = x0.w;
  a.u[4] = x1.x; a.u[5] = x1.y; a.u[6] = x1.z; a.u[7] = x1.w;
  return a;
}

// B-matrix 32x16 bf16 tile from K-pair-packed dwords in global memory.
__device__ __forceinline__ ABu loadB16(const uint32_t* packed, int kk0, int ldn,
                                       int col, int lane) {
  ABu b;
  int kof = (lane >= 16) ? 4 : 0;
#pragma unroll
  for (int v = 0; v < 8; ++v) b.u[v] = packed[(size_t)(kk0 + kof + v + ((v >= 4) ? 4 : 0)) * ldn + col];
  return b;
}

// B-matrix tile from LDS staging buffer (padded rows).
__device__ __forceinline__ ABu loadB16_lds(const uint32_t* lb, int col, int lane) {
  ABu b;
  int kof = (lane >= 16) ? 4 : 0;
#pragma unroll
  for (int v = 0; v < 8; ++v) b.u[v] = lb[(kof + v + ((v >= 4) ? 4 : 0)) * LDS_STRIDE + col];
  return b;
}

__device__ __forceinline__ void async_copy_b128(const uint32_t* g, uint32_t* l) {
#if USE_ASYNC
  __builtin_amdgcn_global_load_async_to_lds_b128((gvec4p)g, (lvec4p)l, 0, 0);
#else
  *(uint4*)l = *(const uint4*)g;
#endif
}

#define WMMA_BF16(A, B, C) \
  __builtin_amdgcn_wmma_f32_16x16x32_bf16(false, (A), false, (B), (short)0, (C), false, false)

// ---------------- elementwise / packing kernels ----------------

__global__ void k_bn0(const float* __restrict__ x, const float* g, const float* b,
                      const float* m, const float* v, unsigned short* __restrict__ xb) {
  int idx = blockIdx.x * blockDim.x + threadIdx.x;
  if (idx >= NN * NF) return;
  int k = idx & (NF - 1);
  float sc = g[k] * __frsqrt_rn(v[k] + EPSB);
  xb[idx] = f2bf((x[idx] - m[k]) * sc + b[k]);
}

__global__ void k_packW(const float* __restrict__ W, uint32_t* __restrict__ WP) {
  int idx = blockIdx.x * blockDim.x + threadIdx.x;          // [NH][64][128]
  if (idx >= NHEADS * 64 * NF) return;
  int n  = idx & (NF - 1);
  int kk = (idx >> 7) & 63;
  int h  = idx >> 13;
  const float* base = W + ((size_t)h * NF + 2 * kk) * HIDN + n;
  WP[idx] = pack2(base[0], base[HIDN]);
}

__global__ void k_packWh(const float* __restrict__ Whf, uint32_t* __restrict__ WhP) {
  int idx = blockIdx.x * blockDim.x + threadIdx.x;          // [NH][3072][128]
  if (idx >= NHEADS * (NN / 2) * HIDN) return;
  int n  = idx & (HIDN - 1);
  int kk = (idx >> 7) % (NN / 2);
  int h  = idx / ((NN / 2) * HIDN);
  const float* base = Whf + ((size_t)h * NN + 2 * kk) * HIDN + n;
  WhP[idx] = pack2(base[0], base[HIDN]);
}

__global__ void k_packGw(const float* __restrict__ gw, uint32_t* __restrict__ gwP) {
  int idx = blockIdx.x * blockDim.x + threadIdx.x;          // [512][48]
  if (idx >= (XC / 2) * NCP) return;
  int n = idx % NCP, kk = idx / NCP;
  float a = (n < NCLS) ? gw[(size_t)(2 * kk) * NCLS + n] : 0.f;
  float b = (n < NCLS) ? gw[(size_t)(2 * kk + 1) * NCLS + n] : 0.f;
  gwP[idx] = pack2(a, b);
}

__global__ void k_packSup(const float* __restrict__ sup, uint32_t* __restrict__ supP) {
  int idx = blockIdx.x * blockDim.x + threadIdx.x;          // [3072][48]
  if (idx >= (NN / 2) * NCP) return;
  int n = idx % NCP, kk = idx / NCP;
  supP[idx] = pack2(sup[(size_t)(2 * kk) * NCP + n], sup[(size_t)(2 * kk + 1) * NCP + n]);
}

// ---------------- Wh = bn0(x) @ W[h]  (WMMA) ----------------

__global__ void k_gemm_wh(const unsigned short* __restrict__ xb,
                          const uint32_t* __restrict__ WP, float* __restrict__ Whf) {
  int wave = (blockIdx.x * blockDim.x + threadIdx.x) >> 5;
  int lane = threadIdx.x & 31;
  int h = wave / (NN / 16), rb = wave % (NN / 16);
  int row  = rb * 16 + (lane & 15);
  const uint32_t* wp = WP + (size_t)h * 64 * NF;
  f32x8 acc[8];
#pragma unroll
  for (int c = 0; c < 8; ++c)
#pragma unroll
    for (int e = 0; e < 8; ++e) acc[c][e] = 0.f;
#pragma unroll
  for (int kt = 0; kt < NF; kt += 32) {
    ABu a = loadA16(xb, row, NF, kt, lane);
#pragma unroll
    for (int c = 0; c < 8; ++c) {
      ABu b = loadB16(wp, kt >> 1, HIDN, c * 16 + (lane & 15), lane);
      acc[c] = WMMA_BF16(a.v, b.v, acc[c]);
    }
  }
  int mrow = rb * 16 + ((lane >= 16) ? 8 : 0);
  int colb = lane & 15;
#pragma unroll
  for (int c = 0; c < 8; ++c)
#pragma unroll
    for (int r = 0; r < 8; ++r)
      Whf[((size_t)h * NN + mrow + r) * HIDN + c * 16 + colb] = acc[c][r];
}

// ---------------- s1/s2 attention dots ----------------

__global__ void k_s12(const float* __restrict__ Whf, const float* __restrict__ asrc,
                      const float* __restrict__ adst, float* __restrict__ s1,
                      float* __restrict__ s2, float* __restrict__ s2t) {
  int idx = blockIdx.x * blockDim.x + threadIdx.x;
  if (idx >= NHEADS * NN) return;
  int h = idx / NN, i = idx % NN;
  const float* wr = Whf + ((size_t)h * NN + i) * HIDN;
  float d1 = 0.f, d2 = 0.f;
#pragma unroll 4
  for (int k = 0; k < HIDN; ++k) {
    float w = wr[k];
    d1 += w * asrc[h * HIDN + k];
    d2 += w * adst[h * HIDN + k];
  }
  s1[h * NN + i] = d1;
  s2[h * NN + i] = d2;
  s2t[i * NHEADS + h] = d2;
}

// ---------------- per-row masked-softmax max & normalizer (all heads) ----------------

__global__ void k_maxsum(const float* __restrict__ adj, const float* __restrict__ s1,
                         const float* __restrict__ s2t, float* __restrict__ mb,
                         float* __restrict__ lb) {
  int i = blockIdx.x * blockDim.x + threadIdx.x;
  if (i >= NN) return;
  const float* arow = adj + (size_t)i * NN;
  float s1h[NHEADS], mh[NHEADS], lh[NHEADS];
#pragma unroll
  for (int h = 0; h < NHEADS; ++h) { s1h[h] = s1[h * NN + i]; mh[h] = -3.0e38f; lh[h] = 0.f; }
  for (int j = 0; j < NN; ++j) {
    if (arow[j] > 0.f) {
      const float* s2r = s2t + (size_t)j * NHEADS;
#pragma unroll
      for (int h = 0; h < NHEADS; ++h) {
        float t = s1h[h] + s2r[h];
        float e = (t > 0.f) ? t : ALPHA * t;
        mh[h] = fmaxf(mh[h], e);
      }
    }
  }
  for (int j = 0; j < NN; ++j) {
    if (arow[j] > 0.f) {
      const float* s2r = s2t + (size_t)j * NHEADS;
#pragma unroll
      for (int h = 0; h < NHEADS; ++h) {
        float t = s1h[h] + s2r[h];
        float e = (t > 0.f) ? t : ALPHA * t;
        lh[h] += __expf(e - mh[h]);
      }
    }
  }
#pragma unroll
  for (int h = 0; h < NHEADS; ++h) { mb[h * NN + i] = mh[h]; lb[h * NN + i] = lh[h]; }
}

// ---------------- flash-style attention aggregation + bn1 + relu (WMMA) ----------------
// Block = 8 waves, ALL on the same head, consecutive 16-row blocks (128 rows).
// Wh B-tiles (shared by the 8 waves) are double-buffer staged into LDS with
// async global->LDS copies; A tiles (normalized probabilities) built in-register.

__global__ void __launch_bounds__(256) k_attn(
    const float* __restrict__ adj, const float* __restrict__ s1,
    const float* __restrict__ s2, const float* __restrict__ mb,
    const float* __restrict__ lb, const uint32_t* __restrict__ WhP,
    const float* g1, const float* b1, const float* m1, const float* v1,
    unsigned short* __restrict__ xcat) {
  __shared__ uint32_t ldsb[2][16 * LDS_STRIDE];

  int tid  = threadIdx.x;
  int wv   = tid >> 5;
  int lane = tid & 31;
  int h    = blockIdx.x / (NN / 16 / 8);   // 48 row-groups per head
  int rbg  = blockIdx.x % (NN / 16 / 8);
  int rb   = rbg * 8 + wv;
  int row  = rb * 16 + (lane & 15);
  int koff = (lane >= 16) ? 8 : 0;

  float s1r  = s1[h * NN + row];
  float mv   = mb[h * NN + row];
  float rinv = 1.f / lb[h * NN + row];
  const uint32_t* whp  = WhP + (size_t)h * (NN / 2) * HIDN;
  const float*    arow = adj + (size_t)row * NN;
  const float*    s2r  = s2 + (size_t)h * NN;

  f32x8 acc[8];
#pragma unroll
  for (int c = 0; c < 8; ++c)
#pragma unroll
    for (int e = 0; e < 8; ++e) acc[c][e] = 0.f;

  // prologue: stage first B tile (16 packed rows x 128 dwords) into buffer 0
  int sr = tid >> 3, ss = tid & 7;           // staging row / 16B segment
  if (tid < 128)
    async_copy_b128(whp + (size_t)0 * HIDN + sr * HIDN + ss * 4,
                    &ldsb[0][sr * LDS_STRIDE + ss * 4]);

  for (int jt = 0; jt < NN / 32; ++jt) {
    int j0 = jt * 32;
#if USE_ASYNC
    __builtin_amdgcn_s_wait_asynccnt(0);
#endif
    __syncthreads();
    if (jt + 1 < NN / 32 && tid < 128)
      async_copy_b128(whp + (size_t)((j0 + 32) >> 1) * HIDN + sr * HIDN + ss * 4,
                      &ldsb[(jt + 1) & 1][sr * LDS_STRIDE + ss * 4]);

    // build normalized-probability A tile for this wave's 16 rows
    ABu a;
#pragma unroll
    for (int seg = 0; seg < 2; ++seg) {
      int jb = j0 + koff + seg * 16;
      float4 a0 = *(const float4*)(arow + jb);
      float4 a1 = *(const float4*)(arow + jb + 4);
      float4 q0 = *(const float4*)(s2r + jb);
      float4 q1 = *(const float4*)(s2r + jb + 4);
      float av[8] = {a0.x, a0.y, a0.z, a0.w, a1.x, a1.y, a1.z, a1.w};
      float sv[8] = {q0.x, q0.y, q0.z, q0.w, q1.x, q1.y, q1.z, q1.w};
      float pv[8];
#pragma unroll
      for (int t = 0; t < 8; ++t) {
        float e = s1r + sv[t];
        e = (e > 0.f) ? e : ALPHA * e;
        pv[t] = (av[t] > 0.f) ? __expf(e - mv) * rinv : 0.f;
      }
#pragma unroll
      for (int t = 0; t < 4; ++t) a.u[seg * 4 + t] = pack2(pv[2 * t], pv[2 * t + 1]);
    }

    const uint32_t* lbuf = &ldsb[jt & 1][0];
#pragma unroll
    for (int c = 0; c < 8; ++c) {
      ABu b = loadB16_lds(lbuf, c * 16 + (lane & 15), lane);
      acc[c] = WMMA_BF16(a.v, b.v, acc[c]);
    }
  }

  // epilogue: bn1 (eval) + relu, write into xcat (bf16 row-major [NN][1024])
  int mrow = rb * 16 + ((lane >= 16) ? 8 : 0);
  int colb = lane & 15;
#pragma unroll
  for (int c = 0; c < 8; ++c) {
    int col = c * 16 + colb;
    float sc = g1[col] * __frsqrt_rn(v1[col] + EPSB);
    float mm = m1[col], bb = b1[col];
#pragma unroll
    for (int r = 0; r < 8; ++r) {
      float val = (acc[c][r] - mm) * sc + bb;
      val = (val > 0.f) ? val : 0.f;
      xcat[(size_t)(mrow + r) * XC + h * HIDN + col] = f2bf(val);
    }
  }
}

// ---------------- support = xcat @ gc_w  (WMMA, N padded to 48) ----------------

__global__ void k_gemm_sup(const unsigned short* __restrict__ xcat,
                           const uint32_t* __restrict__ gwP, float* __restrict__ sup) {
  int wave = (blockIdx.x * blockDim.x + threadIdx.x) >> 5;
  int lane = threadIdx.x & 31;
  int rb = wave;
  int row = rb * 16 + (lane & 15);
  f32x8 acc[3];
#pragma unroll
  for (int c = 0; c < 3; ++c)
#pragma unroll
    for (int e = 0; e < 8; ++e) acc[c][e] = 0.f;
  for (int kt = 0; kt < XC; kt += 32) {
    ABu a = loadA16(xcat, row, XC, kt, lane);
#pragma unroll
    for (int c = 0; c < 3; ++c) {
      ABu b = loadB16(gwP, kt >> 1, NCP, c * 16 + (lane & 15), lane);
      acc[c] = WMMA_BF16(a.v, b.v, acc[c]);
    }
  }
  int mrow = rb * 16 + ((lane >= 16) ? 8 : 0);
  int colb = lane & 15;
#pragma unroll
  for (int c = 0; c < 3; ++c)
#pragma unroll
    for (int r = 0; r < 8; ++r)
      sup[(size_t)(mrow + r) * NCP + c * 16 + colb] = acc[c][r];
}

// ---------------- g = adj @ support  (WMMA, adj cast bf16 on the fly) ----------------

__global__ void k_gemm_adjsup(const float* __restrict__ adj,
                              const uint32_t* __restrict__ supP, float* __restrict__ gout) {
  int wave = (blockIdx.x * blockDim.x + threadIdx.x) >> 5;
  int lane = threadIdx.x & 31;
  int rb = wave;
  int row = rb * 16 + (lane & 15);
  int koff = (lane >= 16) ? 8 : 0;
  const float* arow = adj + (size_t)row * NN;
  f32x8 acc[3];
#pragma unroll
  for (int c = 0; c < 3; ++c)
#pragma unroll
    for (int e = 0; e < 8; ++e) acc[c][e] = 0.f;
  for (int j0 = 0; j0 < NN; j0 += 32) {
    ABu a;
#pragma unroll
    for (int seg = 0; seg < 2; ++seg) {
      int jb = j0 + koff + seg * 16;
      float4 a0 = *(const float4*)(arow + jb);
      float4 a1 = *(const float4*)(arow + jb + 4);
      a.u[seg * 4 + 0] = pack2(a0.x, a0.y);
      a.u[seg * 4 + 1] = pack2(a0.z, a0.w);
      a.u[seg * 4 + 2] = pack2(a1.x, a1.y);
      a.u[seg * 4 + 3] = pack2(a1.z, a1.w);
    }
#pragma unroll
    for (int c = 0; c < 3; ++c) {
      ABu b = loadB16(supP, j0 >> 1, NCP, c * 16 + (lane & 15), lane);
      acc[c] = WMMA_BF16(a.v, b.v, acc[c]);
    }
  }
  int mrow = rb * 16 + ((lane >= 16) ? 8 : 0);
  int colb = lane & 15;
#pragma unroll
  for (int c = 0; c < 3; ++c)
#pragma unroll
    for (int r = 0; r < 8; ++r)
      gout[(size_t)(mrow + r) * NCP + c * 16 + colb] = acc[c][r];
}

// ---------------- final combine + log_softmax ----------------

__global__ void k_final(const float* __restrict__ gbuf, const float* __restrict__ sup,
                        const float* __restrict__ gcb, float* __restrict__ out) {
  int i = blockIdx.x * blockDim.x + threadIdx.x;
  if (i >= NN) return;
  const float* gr = gbuf + (size_t)i * NCP;
  const float* sr = sup + (size_t)i * NCP;
  float vmax = -3.0e38f;
  for (int c = 0; c < NCLS; ++c) {
    float v = (SMOO * gr[c] + sr[c]) / (1.f + SMOO) + gcb[c];
    vmax = fmaxf(vmax, v);
  }
  float s = 0.f;
  for (int c = 0; c < NCLS; ++c) {
    float v = (SMOO * gr[c] + sr[c]) / (1.f + SMOO) + gcb[c];
    s += __expf(v - vmax);
  }
  float lse = vmax + __logf(s);
  for (int c = 0; c < NCLS; ++c) {
    float v = (SMOO * gr[c] + sr[c]) / (1.f + SMOO) + gcb[c];
    out[(size_t)i * NCLS + c] = v - lse;
  }
}

// ---------------- launch ----------------

extern "C" void kernel_launch(void* const* d_in, const int* in_sizes, int n_in,
                              void* d_out, int out_size, void* d_ws, size_t ws_size,
                              hipStream_t stream) {
  (void)in_sizes; (void)n_in; (void)out_size; (void)ws_size;
  const float* x     = (const float*)d_in[0];
  const float* adj   = (const float*)d_in[1];
  const float* W_att = (const float*)d_in[2];
  const float* a_src = (const float*)d_in[3];
  const float* a_dst = (const float*)d_in[4];
  const float* bn0g  = (const float*)d_in[5];
  const float* bn0b  = (const float*)d_in[6];
  const float* bn0m  = (const float*)d_in[7];
  const float* bn0v  = (const float*)d_in[8];
  const float* bn1g  = (const float*)d_in[9];
  const float* bn1b  = (const float*)d_in[10];
  const float* bn1m  = (const float*)d_in[11];
  const float* bn1v  = (const float*)d_in[12];
  const float* gc_w  = (const float*)d_in[13];
  const float* gc_b  = (const float*)d_in[14];
  float* out = (float*)d_out;

  char* w = (char*)d_ws;
  size_t off = 0;
  auto alloc = [&](size_t bytes) -> void* {
    void* p = w + off;
    off = (off + bytes + 255) & ~(size_t)255;
    return p;
  };
  unsigned short* xb   = (unsigned short*)alloc((size_t)NN * NF * 2);
  uint32_t*       WP   = (uint32_t*)alloc((size_t)NHEADS * 64 * NF * 4);
  float*          Whf  = (float*)alloc((size_t)NHEADS * NN * HIDN * 4);
  uint32_t*       WhP  = (uint32_t*)alloc((size_t)NHEADS * (NN / 2) * HIDN * 4);
  float*          s1   = (float*)alloc((size_t)NHEADS * NN * 4);
  float*          s2   = (float*)alloc((size_t)NHEADS * NN * 4);
  float*          s2t  = (float*)alloc((size_t)NN * NHEADS * 4);
  float*          mb   = (float*)alloc((size_t)NHEADS * NN * 4);
  float*          lb   = (float*)alloc((size_t)NHEADS * NN * 4);
  unsigned short* xcat = (unsigned short*)alloc((size_t)NN * XC * 2);
  uint32_t*       gwP  = (uint32_t*)alloc((size_t)(XC / 2) * NCP * 4);
  float*          sup  = (float*)alloc((size_t)NN * NCP * 4);
  uint32_t*       supP = (uint32_t*)alloc((size_t)(NN / 2) * NCP * 4);
  float*          gbuf = (float*)alloc((size_t)NN * NCP * 4);

  const int B = 256;
  k_bn0<<<(NN * NF + B - 1) / B, B, 0, stream>>>(x, bn0g, bn0b, bn0m, bn0v, xb);
  k_packW<<<(NHEADS * 64 * NF + B - 1) / B, B, 0, stream>>>(W_att, WP);
  k_gemm_wh<<<(NHEADS * (NN / 16) * 32) / B, B, 0, stream>>>(xb, WP, Whf);
  k_s12<<<(NHEADS * NN + B - 1) / B, B, 0, stream>>>(Whf, a_src, a_dst, s1, s2, s2t);
  k_packWh<<<(NHEADS * (NN / 2) * HIDN + B - 1) / B, B, 0, stream>>>(Whf, WhP);
  k_maxsum<<<(NN + B - 1) / B, B, 0, stream>>>(adj, s1, s2t, mb, lb);
  k_attn<<<NHEADS * (NN / 16 / 8), B, 0, stream>>>(adj, s1, s2, mb, lb, WhP,
                                                   bn1g, bn1b, bn1m, bn1v, xcat);
  k_packGw<<<((XC / 2) * NCP + B - 1) / B, B, 0, stream>>>(gc_w, gwP);
  k_gemm_sup<<<((NN / 16) * 32) / B, B, 0, stream>>>(xcat, gwP, sup);
  k_packSup<<<((NN / 2) * NCP + B - 1) / B, B, 0, stream>>>(sup, supP);
  k_gemm_adjsup<<<((NN / 16) * 32) / B, B, 0, stream>>>(adj, supP, gbuf);
  k_final<<<(NN + B - 1) / B, B, 0, stream>>>(gbuf, sup, gc_b, out);
}
